// PointTransformerCls_75093208203624
// MI455X (gfx1250) — compile-verified, matrix operand order checked by reading, not run on hardware
//
#include <hip/hip_runtime.h>
#include <hip/hip_bf16.h>

typedef __attribute__((ext_vector_type(16))) _Float16 v16h;
typedef __attribute__((ext_vector_type(8)))  _Float16 v8h;
typedef __attribute__((ext_vector_type(8)))  float    v8f;
typedef int v4i_vs __attribute__((vector_size(16)));   // matches async-LDS builtin param type

#define EPSBN 1e-5f

#if defined(__HIP_DEVICE_COMPILE__) && \
    __has_builtin(__builtin_amdgcn_global_load_async_to_lds_b128) && \
    __has_builtin(__builtin_amdgcn_s_wait_asynccnt)
#define USE_ASYNC_LDS 1
#else
#define USE_ASYNC_LDS 0
#endif

// ---------------------------------------------------------------------------
// Weight pack: W[Kin,Nin] f32 row-major -> f16 WMMA B-matrix tile layout
// pack[((kt*Ntiles+nt)*32 + lane)*16 + e], K(e) = kt*32 + (e<8 ? 8h+e : 16+8h+e-8)
// ---------------------------------------------------------------------------
__global__ void pack_w_kernel(const float* __restrict__ W, _Float16* __restrict__ out,
                              int Kin, int Nin, int Ktiles, int Ntiles) {
    int t = blockIdx.x * blockDim.x + threadIdx.x;
    int total = Ktiles * Ntiles * 32;
    if (t >= total) return;
    int lane = t & 31;
    int nt   = (t >> 5) % Ntiles;
    int kt   = (t >> 5) / Ntiles;
    int h    = lane >> 4;
    int n    = nt * 16 + (lane & 15);
    _Float16* o = out + (size_t)t * 16;
#pragma unroll
    for (int e = 0; e < 16; ++e) {
        int k = kt * 32 + (e < 8 ? 8 * h + e : 16 + 8 * h + (e - 8));
        float v = (k < Kin && n < Nin) ? W[(size_t)k * Nin + n] : 0.0f;
        o[e] = (_Float16)v;
    }
}

__global__ void pad_bias_kernel(const float* __restrict__ in, float* __restrict__ out,
                                int nin, int nout) {
    int i = blockIdx.x * blockDim.x + threadIdx.x;
    if (i < nout) out[i] = (i < nin) ? in[i] : 0.0f;
}

__global__ void f32_to_f16_kernel(const float* __restrict__ in, _Float16* __restrict__ out, int n) {
    int i = blockIdx.x * blockDim.x + threadIdx.x;
    if (i < n) out[i] = (_Float16)in[i];
}

// ---------------------------------------------------------------------------
// enc1 features: [65536, 32] f16 rows = [p0(3), x0(3), 0...]
// ---------------------------------------------------------------------------
__global__ void feat_enc1_kernel(const float* __restrict__ p0, const float* __restrict__ x0,
                                 _Float16* __restrict__ feat, int total) {
    int i = blockIdx.x * blockDim.x + threadIdx.x;
    if (i >= total) return;
    _Float16* fr = feat + (size_t)i * 32;
    fr[0] = (_Float16)p0[3 * i + 0];
    fr[1] = (_Float16)p0[3 * i + 1];
    fr[2] = (_Float16)p0[3 * i + 2];
    fr[3] = (_Float16)x0[3 * i + 0];
    fr[4] = (_Float16)x0[3 * i + 1];
    fr[5] = (_Float16)x0[3 * i + 2];
#pragma unroll
    for (int c = 6; c < 32; ++c) fr[c] = (_Float16)0.0f;
}

// ---------------------------------------------------------------------------
// FPS: one block per batch. LDS-resident coords + distance array.
// Fused min-update + argmax per round; wave32 shuffle reduction.
// dyn smem: 4*n floats (px,py,pz,dist)
// ---------------------------------------------------------------------------
__global__ void fps_kernel(const float* __restrict__ p, int n, int m,
                           int* __restrict__ idx_out, float* __restrict__ p_out) {
    extern __shared__ float smem[];
    float* px = smem;
    float* py = smem + n;
    float* pz = smem + 2 * n;
    float* dist = smem + 3 * n;
    __shared__ float red_v[8];
    __shared__ int   red_i[8];
    __shared__ float selx, sely, selz;

    int b   = blockIdx.x;
    int tid = threadIdx.x;
    const float* pb = p + (size_t)b * n * 3;

    for (int i = tid; i < n; i += blockDim.x) {
        px[i] = pb[3 * i + 0];
        py[i] = pb[3 * i + 1];
        pz[i] = pb[3 * i + 2];
        dist[i] = 3.0e38f;
    }
    if (tid == 0) {
        selx = pb[0]; sely = pb[1]; selz = pb[2];
        idx_out[(size_t)b * m] = 0;
        p_out[((size_t)b * m) * 3 + 0] = pb[0];
        p_out[((size_t)b * m) * 3 + 1] = pb[1];
        p_out[((size_t)b * m) * 3 + 2] = pb[2];
    }
    __syncthreads();

    int nwaves = blockDim.x >> 5;
    for (int s = 1; s < m; ++s) {
        float cx = selx, cy = sely, cz = selz;
        float bv = -1.0f;
        int   bi = 0;
        for (int i = tid; i < n; i += blockDim.x) {
            float dx = px[i] - cx, dy = py[i] - cy, dz = pz[i] - cz;
            float d = fminf(dist[i], dx * dx + dy * dy + dz * dz);
            dist[i] = d;
            if (d > bv) { bv = d; bi = i; }   // strided i increasing -> first-index tie keep
        }
#pragma unroll
        for (int off = 16; off > 0; off >>= 1) {
            float ov = __shfl_down(bv, off, 32);
            int   oi = __shfl_down(bi, off, 32);
            if (ov > bv || (ov == bv && oi < bi)) { bv = ov; bi = oi; }
        }
        int w = tid >> 5;
        if ((tid & 31) == 0) { red_v[w] = bv; red_i[w] = bi; }
        __syncthreads();
        if (tid == 0) {
            for (int w2 = 1; w2 < nwaves; ++w2) {
                if (red_v[w2] > bv || (red_v[w2] == bv && red_i[w2] < bi)) {
                    bv = red_v[w2]; bi = red_i[w2];
                }
            }
            idx_out[(size_t)b * m + s] = bi;
            selx = px[bi]; sely = py[bi]; selz = pz[bi];
            p_out[((size_t)b * m + s) * 3 + 0] = px[bi];
            p_out[((size_t)b * m + s) * 3 + 1] = py[bi];
            p_out[((size_t)b * m + s) * 3 + 2] = pz[bi];
        }
        __syncthreads();
    }
}

// ---------------------------------------------------------------------------
// KNN (k=8): one thread per query; reference points staged to LDS per tile
// via GLOBAL_LOAD_ASYNC_TO_LDS_B128 (ASYNCcnt) when available.
// Tile bytes (cnt*12) are always a multiple of 16 for our sizes.
// grid: (ceil(m/256), B)
// ---------------------------------------------------------------------------
#define KNN_TILE 256
__global__ void knn_kernel(const float* __restrict__ ref, const float* __restrict__ qry,
                           int n, int m, int* __restrict__ knn) {
    __shared__ float tileS[KNN_TILE * 3];
    int b = blockIdx.y;
    int j = blockIdx.x * blockDim.x + threadIdx.x;
    bool act = j < m;
    const float* rb = ref + (size_t)b * n * 3;
    float qx = 0, qy = 0, qz = 0;
    if (act) {
        const float* q = qry + ((size_t)b * m + j) * 3;
        qx = q[0]; qy = q[1]; qz = q[2];
    }
    float bd[8];
    int   bi8[8];
#pragma unroll
    for (int q = 0; q < 8; ++q) { bd[q] = 3.0e38f; bi8[q] = 0; }

    for (int t0 = 0; t0 < n; t0 += KNN_TILE) {
        int cnt = n - t0 < KNN_TILE ? n - t0 : KNN_TILE;
        int bytes = cnt * 12;                       // multiple of 16 for our sizes
        __syncthreads();
#if USE_ASYNC_LDS
        {
            const char* gsrc = (const char*)rb + (size_t)t0 * 12;
            char* ldst = (char*)tileS;
            for (int o = threadIdx.x * 16; o < bytes; o += blockDim.x * 16) {
                __builtin_amdgcn_global_load_async_to_lds_b128(
                    (v4i_vs*)(gsrc + o), (v4i_vs*)(ldst + o), 0, 0);
            }
            __builtin_amdgcn_s_wait_asynccnt(0);
        }
#else
        for (int i = threadIdx.x; i < cnt * 3; i += blockDim.x)
            tileS[i] = rb[(size_t)t0 * 3 + i];
#endif
        __syncthreads();
        if (act) {
            for (int i = 0; i < cnt; ++i) {
                float dx = tileS[3 * i + 0] - qx;
                float dy = tileS[3 * i + 1] - qy;
                float dz = tileS[3 * i + 2] - qz;
                float d = dx * dx + dy * dy + dz * dz;
                if (d < bd[7]) {
                    bd[7] = d; bi8[7] = t0 + i;
#pragma unroll
                    for (int q = 7; q > 0; --q) {
                        if (bd[q] < bd[q - 1]) {
                            float tv = bd[q]; bd[q] = bd[q - 1]; bd[q - 1] = tv;
                            int ti = bi8[q]; bi8[q] = bi8[q - 1]; bi8[q - 1] = ti;
                        }
                    }
                }
            }
        }
    }
    if (act) {
        int* o = knn + ((size_t)b * m + j) * 8;
#pragma unroll
        for (int q = 0; q < 8; ++q) o[q] = bi8[q];
    }
}

// ---------------------------------------------------------------------------
// Grouped feature build: row = q*8 + k, feat = [p[knn]-pn(3), x[knn](C), 0pad]
// ---------------------------------------------------------------------------
__global__ void feat_group_kernel(const float* __restrict__ p, const float* __restrict__ pn,
                                  const float* __restrict__ x, const int* __restrict__ knn,
                                  _Float16* __restrict__ feat,
                                  int n, int m, int C, int Kpad, int total_rows) {
    int row = blockIdx.x * blockDim.x + threadIdx.x;
    if (row >= total_rows) return;
    int q = row >> 3;            // global sample index b*m + j
    int b = q / m;
    int i = knn[row];
    const float* pr = p + ((size_t)b * n + i) * 3;
    const float* pq = pn + (size_t)q * 3;
    _Float16* fr = feat + (size_t)row * Kpad;
    fr[0] = (_Float16)(pr[0] - pq[0]);
    fr[1] = (_Float16)(pr[1] - pq[1]);
    fr[2] = (_Float16)(pr[2] - pq[2]);
    const float* xr = x + ((size_t)b * n + i) * C;
    for (int c = 0; c < C; ++c) fr[3 + c] = (_Float16)xr[c];
    for (int c = 3 + C; c < Kpad; ++c) fr[c] = (_Float16)0.0f;
}

// ---------------------------------------------------------------------------
// WMMA GEMM: one wave computes a 16-row x 2-tile (32-col) strip: A tile loaded
// once per K step feeds two v_wmma_f32_16x16x32_f16. Wave-uniform scalars are
// readfirstlane'd into SGPRs so tile-count control flow is scalar branches
// (no per-iteration EXEC masking around the WMMAs). Fused bias+BN+ReLU epilogue
// and optional neighbor(8) max-pool (a lane's 8 acc VGPRs = the 8 neighbor rows
// of one sample).
// ---------------------------------------------------------------------------
__global__ __launch_bounds__(128)
void wmma_gemm_kernel(const _Float16* __restrict__ A, const _Float16* __restrict__ Bp,
                      float* __restrict__ C, int R, int N, int Kpad,
                      const float* __restrict__ bias, const float* __restrict__ gamma,
                      const float* __restrict__ beta, int do_relu, int do_pool) {
    int wid       = __builtin_amdgcn_readfirstlane(threadIdx.x >> 5);  // SGPR wave id
    int wave      = blockIdx.x * (blockDim.x >> 5) + wid;
    int ntiles    = N >> 4;
    int pairtiles = (ntiles + 1) >> 1;
    int mt  = wave / pairtiles;
    int pr  = wave % pairtiles;
    if (mt * 16 >= R) return;                 // scalar branch
    int nt0  = pr * 2;
    int tcnt = (ntiles - nt0) < 2 ? (ntiles - nt0) : 2;
    int lane = threadIdx.x & 31;
    int h  = lane >> 4;
    int ln = lane & 15;

    v8f acc0 = {};
    v8f acc1 = {};
    int ktiles = Kpad >> 5;
    const _Float16* arow = A + (size_t)(mt * 16 + ln) * Kpad + 8 * h;
    const _Float16* brow = Bp + ((size_t)nt0 * 32 + lane) * 16;
    const size_t bstep = (size_t)ntiles * 512;   // f16 elems per K-tile block

    if (tcnt == 2) {
        for (int kt = 0; kt < ktiles; ++kt) {
            union { v16h v; v8h h8[2]; } ua, ub0, ub1;
            ua.h8[0] = *(const v8h*)(arow);
            ua.h8[1] = *(const v8h*)(arow + 16);
            ub0.h8[0] = *(const v8h*)(brow);
            ub0.h8[1] = *(const v8h*)(brow + 8);
            ub1.h8[0] = *(const v8h*)(brow + 512);
            ub1.h8[1] = *(const v8h*)(brow + 520);
            __builtin_prefetch((const void*)(brow + bstep), 0, 3);
            acc0 = __builtin_amdgcn_wmma_f32_16x16x32_f16(
                false, ua.v, false, ub0.v, (short)0, acc0, false, false);
            acc1 = __builtin_amdgcn_wmma_f32_16x16x32_f16(
                false, ua.v, false, ub1.v, (short)0, acc1, false, false);
            arow += 32;
            brow += bstep;
        }
    } else {
        for (int kt = 0; kt < ktiles; ++kt) {
            union { v16h v; v8h h8[2]; } ua, ub0;
            ua.h8[0] = *(const v8h*)(arow);
            ua.h8[1] = *(const v8h*)(arow + 16);
            ub0.h8[0] = *(const v8h*)(brow);
            ub0.h8[1] = *(const v8h*)(brow + 8);
            __builtin_prefetch((const void*)(brow + bstep), 0, 3);
            acc0 = __builtin_amdgcn_wmma_f32_16x16x32_f16(
                false, ua.v, false, ub0.v, (short)0, acc0, false, false);
            arow += 32;
            brow += bstep;
        }
    }

    float sc = rsqrtf(1.0f + EPSBN);
#pragma unroll
    for (int t = 0; t < 2; ++t) {
        if (t >= tcnt) break;
        int ncol = (nt0 + t) * 16 + ln;
        float bs = bias  ? bias[ncol]  : 0.0f;
        float g  = gamma ? gamma[ncol] : 1.0f;
        float bt = gamma ? beta[ncol]  : 0.0f;
        v8f acc = (t == 0) ? acc0 : acc1;
        float vals[8];
#pragma unroll
        for (int v = 0; v < 8; ++v) {
            float tv = acc[v] + bs;
            if (gamma) tv = g * tv * sc + bt;
            if (do_relu) tv = fmaxf(tv, 0.0f);
            vals[v] = tv;
        }
        if (do_pool) {
            float mx = vals[0];
#pragma unroll
            for (int v = 1; v < 8; ++v) mx = fmaxf(mx, vals[v]);
            int sample = mt * 2 + h;           // rows are sample-major (8 rows/sample)
            C[(size_t)sample * N + ncol] = mx;
        } else {
#pragma unroll
            for (int v = 0; v < 8; ++v)
                C[(size_t)(mt * 16 + v + 8 * h) * N + ncol] = vals[v];
        }
    }
}

// ---------------------------------------------------------------------------
// Segment mean pooling: x5[B*32, 512] -> featc[16, 512] f16 (rows 8..15 zero)
// ---------------------------------------------------------------------------
__global__ void meanpool_kernel(const float* __restrict__ x5, _Float16* __restrict__ featc) {
    int t = blockIdx.x * blockDim.x + threadIdx.x;
    if (t >= 16 * 512) return;
    int col = t % 512;
    int row = t / 512;
    float s = 0.0f;
    if (row < 8) {
        for (int r = 0; r < 32; ++r) s += x5[((size_t)row * 32 + r) * 512 + col];
        s *= (1.0f / 32.0f);
    }
    featc[(size_t)row * 512 + col] = (_Float16)s;
}

__global__ void final_copy_kernel(const float* __restrict__ y, float* __restrict__ out) {
    int t = blockIdx.x * blockDim.x + threadIdx.x;
    if (t >= 8 * 40) return;
    int b = t / 40, c = t % 40;
    out[t] = y[(size_t)b * 48 + c];
}

// ---------------------------------------------------------------------------
// Host orchestration
// ---------------------------------------------------------------------------
static inline int cdiv(int a, int b) { return (a + b - 1) / b; }

extern "C" void kernel_launch(void* const* d_in, const int* in_sizes, int n_in,
                              void* d_out, int out_size, void* d_ws, size_t ws_size,
                              hipStream_t stream) {
    (void)in_sizes; (void)n_in; (void)out_size; (void)ws_size;
    const float* p0  = (const float*)d_in[0];
    const float* x0  = (const float*)d_in[1];
    const float* W1  = (const float*)d_in[3];
    const float* g1  = (const float*)d_in[4];
    const float* b1  = (const float*)d_in[5];
    const float* W2  = (const float*)d_in[6];
    const float* g2  = (const float*)d_in[7];
    const float* b2  = (const float*)d_in[8];
    const float* W3  = (const float*)d_in[9];
    const float* g3  = (const float*)d_in[10];
    const float* b3  = (const float*)d_in[11];
    const float* W4  = (const float*)d_in[12];
    const float* g4  = (const float*)d_in[13];
    const float* b4  = (const float*)d_in[14];
    const float* W5  = (const float*)d_in[15];
    const float* g5  = (const float*)d_in[16];
    const float* b5  = (const float*)d_in[17];
    const float* Wc1 = (const float*)d_in[18];
    const float* bc1 = (const float*)d_in[19];
    const float* gcl = (const float*)d_in[20];
    const float* bcl = (const float*)d_in[21];
    const float* Wc2 = (const float*)d_in[22];
    const float* bc2 = (const float*)d_in[23];
    float* out = (float*)d_out;

    const int B = 8, N0 = 8192;
    const int TOT0 = B * N0;                      // 65536
    const int m2 = 2048, m3 = 512, m4 = 128, m5 = 32;

    // deterministic workspace carve (256B aligned)
    char* wsb = (char*)d_ws;
    size_t off = 0;
    auto alloc = [&](size_t bytes) -> void* {
        void* r = wsb + off;
        off = (off + bytes + 255) & ~(size_t)255;
        return r;
    };
    _Float16* feat1 = (_Float16*)alloc((size_t)TOT0 * 32 * 2);
    float*    x1    = (float*)   alloc((size_t)TOT0 * 32 * 4);
    float*    p2    = (float*)   alloc((size_t)B * m2 * 3 * 4);
    int*      idx2  = (int*)     alloc((size_t)B * m2 * 4);
    int*      knn2  = (int*)     alloc((size_t)B * m2 * 8 * 4);
    _Float16* feat2 = (_Float16*)alloc((size_t)B * m2 * 8 * 64 * 2);
    float*    x2    = (float*)   alloc((size_t)B * m2 * 64 * 4);
    float*    p3    = (float*)   alloc((size_t)B * m3 * 3 * 4);
    int*      idx3  = (int*)     alloc((size_t)B * m3 * 4);
    int*      knn3  = (int*)     alloc((size_t)B * m3 * 8 * 4);
    _Float16* feat3 = (_Float16*)alloc((size_t)B * m3 * 8 * 96 * 2);
    float*    x3    = (float*)   alloc((size_t)B * m3 * 128 * 4);
    float*    p4    = (float*)   alloc((size_t)B * m4 * 3 * 4);
    int*      idx4  = (int*)     alloc((size_t)B * m4 * 4);
    int*      knn4  = (int*)     alloc((size_t)B * m4 * 8 * 4);
    _Float16* feat4 = (_Float16*)alloc((size_t)B * m4 * 8 * 160 * 2);
    float*    x4    = (float*)   alloc((size_t)B * m4 * 256 * 4);
    float*    p5    = (float*)   alloc((size_t)B * m5 * 3 * 4);
    int*      idx5  = (int*)     alloc((size_t)B * m5 * 4);
    int*      knn5  = (int*)     alloc((size_t)B * m5 * 8 * 4);
    _Float16* feat5 = (_Float16*)alloc((size_t)B * m5 * 8 * 288 * 2);
    float*    x5    = (float*)   alloc((size_t)B * m5 * 512 * 4);
    _Float16* W1p   = (_Float16*)alloc((size_t)32 * 32 * 2);
    _Float16* W2p   = (_Float16*)alloc((size_t)64 * 64 * 2);
    _Float16* W3p   = (_Float16*)alloc((size_t)96 * 128 * 2);
    _Float16* W4p   = (_Float16*)alloc((size_t)160 * 256 * 2);
    _Float16* W5p   = (_Float16*)alloc((size_t)288 * 512 * 2);
    _Float16* Wc1p  = (_Float16*)alloc((size_t)512 * 256 * 2);
    _Float16* Wc2p  = (_Float16*)alloc((size_t)256 * 48 * 2);
    _Float16* featc1= (_Float16*)alloc((size_t)16 * 512 * 2);
    float*    yc1   = (float*)   alloc((size_t)16 * 256 * 4);
    _Float16* featc2= (_Float16*)alloc((size_t)16 * 256 * 2);
    float*    yout  = (float*)   alloc((size_t)16 * 48 * 4);
    float*    bc2p  = (float*)   alloc((size_t)48 * 4);

    // -- pack all weights into WMMA B-matrix layout --
    auto pack = [&](const float* W, _Float16* o, int Kin, int Nin, int Ktiles, int Ntiles) {
        int tot = Ktiles * Ntiles * 32;
        pack_w_kernel<<<cdiv(tot, 128), 128, 0, stream>>>(W, o, Kin, Nin, Ktiles, Ntiles);
    };
    pack(W1, W1p, 6, 32, 1, 2);
    pack(W2, W2p, 35, 64, 2, 4);
    pack(W3, W3p, 67, 128, 3, 8);
    pack(W4, W4p, 131, 256, 5, 16);
    pack(W5, W5p, 259, 512, 9, 32);
    pack(Wc1, Wc1p, 512, 256, 16, 16);
    pack(Wc2, Wc2p, 256, 40, 8, 3);
    pad_bias_kernel<<<1, 64, 0, stream>>>(bc2, bc2p, 40, 48);

    auto gemm = [&](const _Float16* A, const _Float16* Bp, float* C, int R, int Ncols,
                    int Kpad, const float* bias, const float* gam, const float* bet,
                    int relu, int pool) {
        int ntiles = Ncols / 16;
        int waves = (R / 16) * cdiv(ntiles, 2);
        wmma_gemm_kernel<<<cdiv(waves, 4), 128, 0, stream>>>(A, Bp, C, R, Ncols, Kpad,
                                                             bias, gam, bet, relu, pool);
    };

    // -- enc1 --
    feat_enc1_kernel<<<cdiv(TOT0, 256), 256, 0, stream>>>(p0, x0, feat1, TOT0);
    gemm(feat1, W1p, x1, TOT0, 32, 32, nullptr, g1, b1, 1, 0);

    // -- stage 2: 8192 -> 2048, C=32 -> 64 --
    fps_kernel<<<B, 256, 16 * N0, stream>>>(p0, N0, m2, idx2, p2);
    { dim3 g(cdiv(m2, 256), B); knn_kernel<<<g, 256, 0, stream>>>(p0, p2, N0, m2, knn2); }
    feat_group_kernel<<<cdiv(B * m2 * 8, 256), 256, 0, stream>>>(p0, p2, x1, knn2, feat2,
                                                                 N0, m2, 32, 64, B * m2 * 8);
    gemm(feat2, W2p, x2, B * m2 * 8, 64, 64, nullptr, g2, b2, 1, 1);

    // -- stage 3: 2048 -> 512, C=64 -> 128 --
    fps_kernel<<<B, 256, 16 * m2, stream>>>(p2, m2, m3, idx3, p3);
    { dim3 g(cdiv(m3, 256), B); knn_kernel<<<g, 256, 0, stream>>>(p2, p3, m2, m3, knn3); }
    feat_group_kernel<<<cdiv(B * m3 * 8, 256), 256, 0, stream>>>(p2, p3, x2, knn3, feat3,
                                                                 m2, m3, 64, 96, B * m3 * 8);
    gemm(feat3, W3p, x3, B * m3 * 8, 128, 96, nullptr, g3, b3, 1, 1);

    // -- stage 4: 512 -> 128, C=128 -> 256 --
    fps_kernel<<<B, 256, 16 * m3, stream>>>(p3, m3, m4, idx4, p4);
    { dim3 g(cdiv(m4, 256), B); knn_kernel<<<g, 256, 0, stream>>>(p3, p4, m3, m4, knn4); }
    feat_group_kernel<<<cdiv(B * m4 * 8, 256), 256, 0, stream>>>(p3, p4, x3, knn4, feat4,
                                                                 m3, m4, 128, 160, B * m4 * 8);
    gemm(feat4, W4p, x4, B * m4 * 8, 256, 160, nullptr, g4, b4, 1, 1);

    // -- stage 5: 128 -> 32, C=256 -> 512 --
    fps_kernel<<<B, 256, 16 * m4, stream>>>(p4, m4, m5, idx5, p5);
    { dim3 g(cdiv(m5, 256), B); knn_kernel<<<g, 256, 0, stream>>>(p4, p5, m4, m5, knn5); }
    feat_group_kernel<<<cdiv(B * m5 * 8, 256), 256, 0, stream>>>(p4, p5, x4, knn5, feat5,
                                                                 m4, m5, 256, 288, B * m5 * 8);
    gemm(feat5, W5p, x5, B * m5 * 8, 512, 288, nullptr, g5, b5, 1, 1);

    // -- classifier --
    meanpool_kernel<<<cdiv(16 * 512, 256), 256, 0, stream>>>(x5, featc1);
    gemm(featc1, Wc1p, yc1, 16, 256, 512, bc1, gcl, bcl, 1, 0);
    f32_to_f16_kernel<<<cdiv(16 * 256, 256), 256, 0, stream>>>(yc1, featc2, 16 * 256);
    gemm(featc2, Wc2p, yout, 16, 48, 256, bc2p, nullptr, nullptr, 0, 0);
    final_copy_kernel<<<2, 256, 0, stream>>>(yout, out);
}
